// StreamingFCBlock_65884798320785
// MI455X (gfx1250) — matrix operand order repaired and checked by reading, not compiled, not run on hardware
//
#include <hip/hip_runtime.h>
#include <stdint.h>

#define LN_EPS 1e-5f
#define BN_EPS 1e-5f

// ---------------------------------------------------------------- types
typedef __attribute__((ext_vector_type(16))) __bf16 bf16x16;
typedef __attribute__((ext_vector_type(8)))  float  f32x8;

union FragBF { bf16x16 v; uint4 q[2]; };

__device__ __forceinline__ unsigned short f2bf(float f) {
  unsigned int u = __float_as_uint(f);
  u += 0x7FFFu + ((u >> 16) & 1u);          // round-to-nearest-even
  return (unsigned short)(u >> 16);
}

__device__ __forceinline__ unsigned int pack2bf(float a, float b) {
  return (unsigned int)f2bf(a) | ((unsigned int)f2bf(b) << 16);
}

__device__ __forceinline__ float silu_f(float x) {
  return x * (1.0f / (1.0f + __expf(-x)));
}

// Staging tile registers: pure SSA (returned/passed by value, no unions,
// no address taken) so SROA keeps them in VGPRs, not scratch.
struct TileRegs { uint4 a0, a1, w0, w1; };

template<bool AF32>
__device__ __forceinline__ TileRegs tile_gload(const void* Aptr,
                                               const unsigned short* Wb,
                                               size_t gA, size_t gW) {
  TileRegs t;
  if (AF32) {
    const float* A = (const float*)Aptr;
    float4 f0 = *(const float4*)(A + gA);
    float4 f1 = *(const float4*)(A + gA + 4);
    float4 f2 = *(const float4*)(A + gA + 8);
    float4 f3 = *(const float4*)(A + gA + 12);
    t.a0 = make_uint4(pack2bf(f0.x, f0.y), pack2bf(f0.z, f0.w),
                      pack2bf(f1.x, f1.y), pack2bf(f1.z, f1.w));
    t.a1 = make_uint4(pack2bf(f2.x, f2.y), pack2bf(f2.z, f2.w),
                      pack2bf(f3.x, f3.y), pack2bf(f3.z, f3.w));
  } else {
    const unsigned short* A = (const unsigned short*)Aptr;
    t.a0 = *(const uint4*)(A + gA);
    t.a1 = *(const uint4*)(A + gA + 8);
  }
  t.w0 = *(const uint4*)(Wb + gW);
  t.w1 = *(const uint4*)(Wb + gW + 8);
  return t;
}

__device__ __forceinline__ void tile_dsstore(unsigned short* da,
                                             unsigned short* dw, TileRegs t) {
  *(uint4*)da       = t.a0;
  *(uint4*)(da + 8) = t.a1;
  *(uint4*)dw       = t.w0;
  *(uint4*)(dw + 8) = t.w1;
}

// ---------------------------------------------------------------- WMMA GEMM
// Y[M,N] = epilogue( A[M,K] @ W[N,K]^T + bias[N] )
// Block tile 128x128, 8 waves as 4(M) x 2(N), wave tile 32x64 (2x4 frags).
// Double-buffered LDS, one barrier per k-step, BK = 32.
// Next-tile global loads are issued AFTER the barrier so they stay in flight
// across the whole WMMA block (HIP __syncthreads fences global too, so
// issuing them before the barrier would drain loadcnt immediately).
// A: bf16 (ushort) or f32 (AF32). W: bf16 (ushort), natural [N][K] layout.
// EPI: 0 = bias, f32 out            1 = bias+SiLU, bf16 out
//      2 = resid + 0.5*v, f32 out   3 = resid + v, f32 out
template<bool AF32, int EPI>
__global__ __launch_bounds__(256, 1) void gemm_wmma_kernel(
    const void* __restrict__ Aptr, const unsigned short* __restrict__ Wb,
    const float* __restrict__ bias, const float* __restrict__ resid,
    void* __restrict__ outp, int M, int N, int K)
{
  const int LSTR = 40;                                   // 32 + 8 pad (80 B rows)
  const int STG  = 128 * 40;                             // elems per stage
  __shared__ __align__(16) unsigned short As[2][128 * 40];  // 20 KB
  __shared__ __align__(16) unsigned short Ws[2][128 * 40];  // 20 KB

  const int tid  = threadIdx.x;
  const int bm   = blockIdx.y * 128;
  const int bn   = blockIdx.x * 128;
  const int wave = tid >> 5, lane = tid & 31;
  const int wm = wave >> 1, wn = wave & 1;               // 4 x 2 wave grid
  const int r = lane & 15, half = lane >> 4;

  f32x8 acc[2][4] = {};

  const int row = tid >> 1, seg = tid & 1;               // 128 rows x 2 segs of 16
  const int kSteps = K >> 5;

  const size_t aBase = (size_t)(bm + row) * K + seg * 16;
  const size_t wBase = (size_t)(bn + row) * K + seg * 16;
  unsigned short* const aSlot = &As[0][row * LSTR + seg * 16];
  unsigned short* const wSlot = &Ws[0][row * LSTR + seg * 16];
  const unsigned short* const aFragBase = &As[0][(wm * 32 + r) * LSTR];
  const unsigned short* const bFragBase = &Ws[0][(wn * 64 + r) * LSTR];

  // prologue: tile 0 -> LDS stage 0
  tile_dsstore(aSlot, wSlot, tile_gload<AF32>(Aptr, Wb, aBase, wBase));
  int stage = 0;

  for (int kt = 0; kt < kSteps; ++kt) {
    const bool haveNext = (kt + 1 < kSteps);
    __syncthreads();                                     // stage[stage] fully written

    TileRegs tn;
    if (haveNext)                                        // issue AFTER barrier:
      tn = tile_gload<AF32>(Aptr, Wb, aBase + (size_t)(kt + 1) * 32,
                            wBase + (size_t)(kt + 1) * 32);  // in flight over WMMAs

    // ---- fragments per ISA layouts (two b128 LDS loads each)
    FragBF a[2], b[4];
    #pragma unroll
    for (int f = 0; f < 2; ++f) {
      const unsigned short* ap = aFragBase + stage * STG + f * 16 * LSTR;
      a[f].q[0] = *(const uint4*)(ap + half * 8);        // K = half*8 .. +8
      a[f].q[1] = *(const uint4*)(ap + 16 + half * 8);   // K = 16+half*8 ..
    }
    #pragma unroll
    for (int f = 0; f < 4; ++f) {
      const unsigned short* bp = bFragBase + stage * STG + f * 16 * LSTR;
      b[f].q[0] = *(const uint4*)(bp + half * 8);        // K = half*8 .. +8
      b[f].q[1] = *(const uint4*)(bp + 16 + half * 8);   // K = 16+half*8 ..
    }
    #pragma unroll
    for (int fm = 0; fm < 2; ++fm)
      #pragma unroll
      for (int fn = 0; fn < 4; ++fn)
        acc[fm][fn] = __builtin_amdgcn_wmma_f32_16x16x32_bf16(
            false, a[fm].v, false, b[fn].v, (short)0, acc[fm][fn], false, false);

    if (haveNext) {                                      // other buffer: no race
      int st = stage ^ 1;
      tile_dsstore(aSlot + st * STG, wSlot + st * STG, tn);
    }
    stage ^= 1;
  }

  // ---- epilogue: D vgpr j -> row half*8+j, col = r
  #pragma unroll
  for (int fm = 0; fm < 2; ++fm) {
    #pragma unroll
    for (int fn = 0; fn < 4; ++fn) {
      #pragma unroll
      for (int j = 0; j < 8; ++j) {
        int grow = bm + wm * 32 + fm * 16 + half * 8 + j;
        int gcol = bn + wn * 64 + fn * 16 + r;
        size_t o = (size_t)grow * N + gcol;
        float v = acc[fm][fn][j] + bias[gcol];
        if (EPI == 0)      ((float*)outp)[o] = v;
        else if (EPI == 1) ((unsigned short*)outp)[o] = f2bf(silu_f(v));
        else if (EPI == 2) ((float*)outp)[o] = resid[o] + 0.5f * v;
        else               ((float*)outp)[o] = resid[o] + v;
      }
    }
  }
}

// ---------------------------------------------------------------- LayerNorm -> bf16
template<bool REMAP>
__global__ __launch_bounds__(256) void ln_kernel(
    const float* __restrict__ x, const float* __restrict__ g,
    const float* __restrict__ bb, unsigned short* __restrict__ out,
    int Tn, int Lpad, int PAD)
{
  const int Dk = 1024;
  int row = blockIdx.x;
  int tid = threadIdx.x;
  const float* xr = x + (size_t)row * Dk;
  float v[4]; float s = 0.f;
  #pragma unroll
  for (int i = 0; i < 4; ++i) { v[i] = xr[tid + i * 256]; s += v[i]; }
  __shared__ float red[256];
  red[tid] = s; __syncthreads();
  for (int st = 128; st > 0; st >>= 1) {
    if (tid < st) red[tid] += red[tid + st];
    __syncthreads();
  }
  float mean = red[0] * (1.0f / 1024.0f);
  __syncthreads();
  float s2 = 0.f;
  #pragma unroll
  for (int i = 0; i < 4; ++i) { float d = v[i] - mean; s2 += d * d; }
  red[tid] = s2; __syncthreads();
  for (int st = 128; st > 0; st >>= 1) {
    if (tid < st) red[tid] += red[tid + st];
    __syncthreads();
  }
  float rstd = rsqrtf(red[0] * (1.0f / 1024.0f) + LN_EPS);
  size_t orow = REMAP ? ((size_t)(row / Tn) * Lpad + PAD + (row % Tn))
                      : (size_t)row;
  unsigned short* op = out + orow * Dk;
  #pragma unroll
  for (int i = 0; i < 4; ++i) {
    int c = tid + i * 256;
    op[c] = f2bf((v[i] - mean) * rstd * g[c] + bb[c]);
  }
}

// ---------------------------------------------------------------- fp32 -> bf16
__global__ void f32_to_bf16_kernel(const float* __restrict__ in,
                                   unsigned short* __restrict__ out, int n) {
  int i = blockIdx.x * 256 + threadIdx.x;
  if (i < n) out[i] = f2bf(in[i]);
}

// conv_buf [B,8,D] f32 -> y_in rows 0..7 of [B,136,D] bf16
__global__ void convbuf_to_yin_kernel(const float* __restrict__ cb,
                                      unsigned short* __restrict__ yin) {
  int i = blockIdx.x * 256 + threadIdx.x;
  const int n = 64 * 8 * 1024;
  if (i >= n) return;
  int c = i & 1023;
  int rr = (i >> 10) & 7;
  int b = i >> 13;
  yin[((size_t)b * 136 + rr) * 1024 + c] = f2bf(cb[i]);
}

// ---------------------------------------------------------------- attention
// one block per (b,h); flash-style online softmax; 198 keys (70 ctx + 128 new)
__global__ __launch_bounds__(128, 1) void attn_kernel(
    const float* __restrict__ Q, const float* __restrict__ Kn,
    const float* __restrict__ Vn, const float* __restrict__ Kc,
    const float* __restrict__ Vc, unsigned short* __restrict__ ctx)
{
  const int b = blockIdx.x >> 4, h = blockIdx.x & 15;
  const int t = threadIdx.x;
  const int T = 128, WLk = 70, DK = 64, TOTK = 198, D = 1024;
  __shared__ float Ks[64][64];
  __shared__ float Vs[64][64];

  float q[64];
  {
    const float* qp = Q + ((size_t)(b * T + t)) * D + h * DK;
    #pragma unroll
    for (int d = 0; d < 64; ++d) q[d] = qp[d];
  }
  float m = -3.0e38f, l = 0.0f;
  float acc[64];
  #pragma unroll
  for (int d = 0; d < 64; ++d) acc[d] = 0.0f;

  for (int c0 = 0; c0 < TOTK; c0 += 64) {
    int cn = TOTK - c0; if (cn > 64) cn = 64;
    __syncthreads();
    for (int i = threadIdx.x; i < cn * 64; i += 128) {
      int kr = c0 + (i >> 6), kd = i & 63;
      float kv, vv;
      if (kr < WLk) {
        size_t o = (((size_t)(b * 16 + h)) * WLk + kr) * 64 + kd;
        kv = Kc[o]; vv = Vc[o];
      } else {
        size_t o = ((size_t)(b * T + (kr - WLk))) * D + h * DK + kd;
        kv = Kn[o]; vv = Vn[o];
      }
      Ks[i >> 6][kd] = kv; Vs[i >> 6][kd] = vv;
    }
    __syncthreads();
    for (int j = 0; j < cn; ++j) {
      float s = 0.0f;
      #pragma unroll
      for (int d = 0; d < 64; ++d) s += q[d] * Ks[j][d];
      s *= 0.125f;                               // 1/sqrt(64)
      if (s > m) {
        float sc = __expf(m - s);
        l *= sc;
        #pragma unroll
        for (int d = 0; d < 64; ++d) acc[d] *= sc;
        m = s;
      }
      float p = __expf(s - m);
      l += p;
      #pragma unroll
      for (int d = 0; d < 64; ++d) acc[d] += p * Vs[j][d];
    }
  }
  float inv = 1.0f / l;
  unsigned short* op = ctx + ((size_t)(b * T + t)) * D + h * DK;
  #pragma unroll
  for (int d = 0; d < 64; ++d) op[d] = f2bf(acc[d] * inv);
}

// ---------------------------------------------------------------- dw-conv + BN + SiLU
// z1 f32 [B,136,2048] -> S bf16 [B,128,2048] (last 128 positions only)
__global__ __launch_bounds__(256) void dwconv_bn_silu_kernel(
    const float* __restrict__ z1, const float* __restrict__ dw_w,
    const float* __restrict__ dw_b, const float* __restrict__ bn_g,
    const float* __restrict__ bn_b, const float* __restrict__ bn_mean,
    const float* __restrict__ bn_var, unsigned short* __restrict__ outS)
{
  const int C = 2048, T = 128, L = 136;
  int idx = blockIdx.x * 256 + threadIdx.x;
  if (idx >= 64 * T * C) return;
  int c = idx % C;
  int t = (idx / C) % T;
  int b = idx / (C * T);
  int l = t + 8;                                 // position in padded stream
  float s = dw_b[c];
  #pragma unroll
  for (int j = 0; j < 9; ++j) {
    int p = l - 4 + j;                           // p >= 4 always
    if (p < L) s += z1[((size_t)b * L + p) * C + c] * dw_w[c * 9 + j];
  }
  float z = (s - bn_mean[c]) * rsqrtf(bn_var[c] + BN_EPS) * bn_g[c] + bn_b[c];
  outS[((size_t)b * T + t) * C + c] = f2bf(silu_f(z));
}

// ---------------------------------------------------------------- launch
extern "C" void kernel_launch(void* const* d_in, const int* in_sizes, int n_in,
                              void* d_out, int out_size, void* d_ws, size_t ws_size,
                              hipStream_t stream) {
  const int B = 64, T = 128, D = 1024, H = 16, TOK = B * T;   // 8192
  (void)in_sizes; (void)n_in; (void)out_size; (void)ws_size;

  const float* x_new   = (const float*)d_in[0];
  const float* K_ctx   = (const float*)d_in[1];
  const float* V_ctx   = (const float*)d_in[2];
  const float* conv_bf = (const float*)d_in[3];
  const float* ln1_g   = (const float*)d_in[4];
  const float* ln1_b   = (const float*)d_in[5];
  const float* fc1_w   = (const float*)d_in[6];
  const float* fc1_b   = (const float*)d_in[7];
  const float* fc2_w   = (const float*)d_in[8];
  const float* fc2_b   = (const float*)d_in[9];
  const float* ln2_g   = (const float*)d_in[10];
  const float* ln2_b   = (const float*)d_in[11];
  const float* q_w = (const float*)d_in[12]; const float* q_b = (const float*)d_in[13];
  const float* k_w = (const float*)d_in[14]; const float* k_b = (const float*)d_in[15];
  const float* v_w = (const float*)d_in[16]; const float* v_b = (const float*)d_in[17];
  const float* o_w = (const float*)d_in[18]; const float* o_b = (const float*)d_in[19];
  const float* ln3_g   = (const float*)d_in[20];
  const float* ln3_b   = (const float*)d_in[21];
  const float* pw1_w   = (const float*)d_in[22];
  const float* pw1_b   = (const float*)d_in[23];
  const float* dw_w    = (const float*)d_in[24];
  const float* dw_b    = (const float*)d_in[25];
  const float* bn_g    = (const float*)d_in[26];
  const float* bn_b    = (const float*)d_in[27];
  const float* bn_mean = (const float*)d_in[28];
  const float* bn_var  = (const float*)d_in[29];
  const float* pw2_w   = (const float*)d_in[30];
  const float* pw2_b   = (const float*)d_in[31];

  // ---- workspace arena (lifetimes reused)
  char* ws = (char*)d_ws;
  size_t off = 0;
  auto arena = [&](size_t bytes) -> void* {
    void* p = ws + off;
    off += (bytes + 255) & ~(size_t)255;
    return p;
  };
  unsigned short* wb_fc1 = (unsigned short*)arena((size_t)4096 * 1024 * 2);
  unsigned short* wb_fc2 = (unsigned short*)arena((size_t)4096 * 1024 * 2);
  unsigned short* wb_q   = (unsigned short*)arena((size_t)1024 * 1024 * 2);
  unsigned short* wb_k   = (unsigned short*)arena((size_t)1024 * 1024 * 2);
  unsigned short* wb_v   = (unsigned short*)arena((size_t)1024 * 1024 * 2);
  unsigned short* wb_o   = (unsigned short*)arena((size_t)1024 * 1024 * 2);
  unsigned short* wb_pw1 = (unsigned short*)arena((size_t)2048 * 1024 * 2);
  unsigned short* wb_pw2 = (unsigned short*)arena((size_t)2048 * 1024 * 2);
  unsigned short* Abuf   = (unsigned short*)arena((size_t)64 * 136 * 1024 * 2); // ln1/ln2/ctx/y_in
  unsigned short* Hbuf   = (unsigned short*)arena((size_t)8192 * 4096 * 2);     // h1 / s / h2
  float* X1 = (float*)arena((size_t)TOK * 1024 * 4);                            // x1 / x3
  float* X2 = (float*)arena((size_t)TOK * 1024 * 4);
  float* Big = (float*)arena((size_t)3 * TOK * 1024 * 4);                       // QKV / z1
  float* Qb = Big;
  float* Kb = Big + (size_t)TOK * 1024;
  float* Vb = Big + (size_t)2 * TOK * 1024;
  float* Z1 = Big;

  // ---- weight conversion fp32 -> bf16
  auto cvt = [&](const float* w, unsigned short* o, int n) {
    f32_to_bf16_kernel<<<(n + 255) / 256, 256, 0, stream>>>(w, o, n);
  };
  cvt(fc1_w, wb_fc1, 4096 * 1024);
  cvt(fc2_w, wb_fc2, 4096 * 1024);
  cvt(q_w, wb_q, 1024 * 1024);
  cvt(k_w, wb_k, 1024 * 1024);
  cvt(v_w, wb_v, 1024 * 1024);
  cvt(o_w, wb_o, 1024 * 1024);
  cvt(pw1_w, wb_pw1, 2048 * 1024);
  cvt(pw2_w, wb_pw2, 2048 * 1024);

  // ---- macaron FFN #1:  x1 = x + 0.5*ffn(LN(x))
  ln_kernel<false><<<TOK, 256, 0, stream>>>(x_new, ln1_g, ln1_b, Abuf, 0, 0, 0);
  gemm_wmma_kernel<false, 1><<<dim3(4096 / 128, TOK / 128), 256, 0, stream>>>(
      Abuf, wb_fc1, fc1_b, nullptr, Hbuf, TOK, 4096, 1024);
  gemm_wmma_kernel<false, 2><<<dim3(1024 / 128, TOK / 128), 256, 0, stream>>>(
      Hbuf, wb_fc2, fc2_b, x_new, X1, TOK, 1024, 4096);

  // ---- attention:  x2 = x1 + O(attn(QKV(LN(x1))))
  ln_kernel<false><<<TOK, 256, 0, stream>>>(X1, ln2_g, ln2_b, Abuf, 0, 0, 0);
  gemm_wmma_kernel<false, 0><<<dim3(1024 / 128, TOK / 128), 256, 0, stream>>>(
      Abuf, wb_q, q_b, nullptr, Qb, TOK, 1024, 1024);
  gemm_wmma_kernel<false, 0><<<dim3(1024 / 128, TOK / 128), 256, 0, stream>>>(
      Abuf, wb_k, k_b, nullptr, Kb, TOK, 1024, 1024);
  gemm_wmma_kernel<false, 0><<<dim3(1024 / 128, TOK / 128), 256, 0, stream>>>(
      Abuf, wb_v, v_b, nullptr, Vb, TOK, 1024, 1024);
  attn_kernel<<<B * H, 128, 0, stream>>>(Qb, Kb, Vb, K_ctx, V_ctx, Abuf);
  gemm_wmma_kernel<false, 3><<<dim3(1024 / 128, TOK / 128), 256, 0, stream>>>(
      Abuf, wb_o, o_b, X1, X2, TOK, 1024, 1024);

  // ---- conv module:  x3 = x2 + pw2(silu(bn(dwconv(pw1(concat(buf, LN(x2)))))))
  convbuf_to_yin_kernel<<<(64 * 8 * 1024 + 255) / 256, 256, 0, stream>>>(conv_bf, Abuf);
  ln_kernel<true><<<TOK, 256, 0, stream>>>(X2, ln3_g, ln3_b, Abuf, T, 136, 8);
  gemm_wmma_kernel<false, 0><<<dim3(2048 / 128, (64 * 136) / 128), 256, 0, stream>>>(
      Abuf, wb_pw1, pw1_b, nullptr, Z1, 64 * 136, 2048, 1024);
  dwconv_bn_silu_kernel<<<(64 * 128 * 2048) / 256, 256, 0, stream>>>(
      Z1, dw_w, dw_b, bn_g, bn_b, bn_mean, bn_var, Hbuf);
  gemm_wmma_kernel<false, 3><<<dim3(1024 / 128, TOK / 128), 256, 0, stream>>>(
      Hbuf, wb_pw2, pw2_b, X2, X1, TOK, 1024, 2048);

  // ---- macaron FFN #2 (shared weights): out = x3 + 0.5*ffn(x3)
  gemm_wmma_kernel<true, 1><<<dim3(4096 / 128, TOK / 128), 256, 0, stream>>>(
      X1, wb_fc1, fc1_b, nullptr, Hbuf, TOK, 4096, 1024);
  gemm_wmma_kernel<false, 2><<<dim3(1024 / 128, TOK / 128), 256, 0, stream>>>(
      Hbuf, wb_fc2, fc2_b, X1, (float*)d_out, TOK, 1024, 4096);
}
